// Generator_1632087573034
// MI455X (gfx1250) — matrix operand order repaired and checked by reading, not compiled, not run on hardware
//
#include <hip/hip_runtime.h>
#include <hip/hip_bf16.h>
#include <math.h>

// Model dims (match reference)
#define B_   64
#define Z_   100
#define L_   1024
#define D_   256
#define CH_  32
#define NL_  4
#define DS_  2
#define DC_  2
#define DI_  256   // expand * D
#define DTR_ 16    // ceil(D/16)

typedef __attribute__((ext_vector_type(16))) _Float16 v16h;
typedef __attribute__((ext_vector_type(8)))  _Float16 v8h;
typedef __attribute__((ext_vector_type(8)))  float    v8f;

#if __has_builtin(__builtin_amdgcn_global_load_async_to_lds_b128)
#define HAVE_ASYNC_LDS 1
// Exact parameter types per hipcc diagnostic: int __vector(4) in AS1 / AS3
typedef int gv4i __attribute__((vector_size(16)));
typedef __attribute__((address_space(1))) gv4i* gv4i_gp;
typedef __attribute__((address_space(3))) gv4i* gv4i_lp;
#else
#define HAVE_ASYNC_LDS 0
#endif

__device__ __forceinline__ float warp_sum32(float v) {
  #pragma unroll
  for (int off = 16; off > 0; off >>= 1) v += __shfl_xor(v, off, 32);
  return v;
}

__device__ __forceinline__ float sigmoidf_(float x) { return 1.f / (1.f + __expf(-x)); }
__device__ __forceinline__ float siluf_(float x)    { return x * sigmoidf_(x); }

__device__ __forceinline__ v16h cat8(v8h lo, v8h hi) {
  v16h r;
  #pragma unroll
  for (int i = 0; i < 8; ++i) { r[i] = lo[i]; r[i + 8] = hi[i]; }
  return r;
}

// Stage one 16-byte segment of the weight tile into LDS.
// CDNA5 path: GLOBAL_LOAD_ASYNC_TO_LDS_B128 (tracked by ASYNCcnt).
// Fallback: plain load + ds_store (covered by the workgroup barrier).
__device__ __forceinline__ void stage_b128(const _Float16* __restrict__ gsrc,
                                           _Float16* __restrict__ ldst) {
#if HAVE_ASYNC_LDS
  __builtin_amdgcn_global_load_async_to_lds_b128(
      (gv4i_gp)(void*)gsrc, (gv4i_lp)(void*)ldst, /*offset=*/0, /*cpol=*/0);
#else
  *(v8h*)ldst = *(const v8h*)gsrc;
#endif
}

// Async loads complete in order: waiting for <=1 outstanding guarantees the
// older (current-buffer) copy has landed while the next-slice copy stays in flight.
__device__ __forceinline__ void wait_async_le1() {
#if HAVE_ASYNC_LDS
#if __has_builtin(__builtin_amdgcn_s_wait_asynccnt)
  __builtin_amdgcn_s_wait_asynccnt(1);
#else
  asm volatile("s_wait_asynccnt 0x1" ::: "memory");
#endif
#endif
}
__device__ __forceinline__ void wait_async_le0() {
#if HAVE_ASYNC_LDS
#if __has_builtin(__builtin_amdgcn_s_wait_asynccnt)
  __builtin_amdgcn_s_wait_asynccnt(0);
#else
  asm volatile("s_wait_asynccnt 0x0" ::: "memory");
#endif
#endif
}

// -------- 1) initial projection: x = noise @ l1_w^T + l1_b + pos_embed --------
__global__ void k_init(const float* __restrict__ noise, const float* __restrict__ w,
                       const float* __restrict__ bvec, const float* __restrict__ pos,
                       float* __restrict__ x) {
  __shared__ float sn[Z_];
  const int b = blockIdx.y;
  const int r = blockIdx.x * 256 + threadIdx.x;   // r in [0, L*D)
  if (threadIdx.x < Z_) sn[threadIdx.x] = noise[b * Z_ + threadIdx.x];
  __syncthreads();
  const float* wr = w + (size_t)r * Z_;
  float acc = bvec[r];
  #pragma unroll 4
  for (int k = 0; k < Z_; ++k) acc = fmaf(sn[k], wr[k], acc);
  x[(size_t)b * (L_ * D_) + r] = acc + pos[r];
}

// -------- fp32 -> f16 weight conversion --------
__global__ void k_cvt_f16(const float* __restrict__ in, _Float16* __restrict__ out, int n) {
  int i = blockIdx.x * 256 + threadIdx.x;
  if (i < n) out[i] = (_Float16)in[i];
}

// -------- layernorm over D=256, one wave32 per row, writes f16 --------
__global__ void k_layernorm(const float* __restrict__ x, const float* __restrict__ w,
                            const float* __restrict__ bb, _Float16* __restrict__ h16) {
  const int row  = blockIdx.x * 8 + (threadIdx.x >> 5);
  const int lane = threadIdx.x & 31;
  const float* xr = x + (size_t)row * D_;
  float v[8];
  float s = 0.f, s2 = 0.f;
  #pragma unroll
  for (int i = 0; i < 8; ++i) {
    float t = xr[lane + i * 32];
    v[i] = t; s += t; s2 = fmaf(t, t, s2);
  }
  s  = warp_sum32(s);
  s2 = warp_sum32(s2);
  const float mu  = s * (1.f / D_);
  const float var = s2 * (1.f / D_) - mu * mu;
  const float inv = rsqrtf(var + 1e-5f);
  _Float16* hr = h16 + (size_t)row * D_;
  #pragma unroll
  for (int i = 0; i < 8; ++i) {
    int d = lane + i * 32;
    hr[d] = (_Float16)((v[i] - mu) * inv * w[d] + bb[d]);
  }
}

// -------- WMMA GEMM: C[M,N] (f32) = A16[M,K] * W16[N,K]^T --------
// block = 256 threads = 8 waves. Block tile = 128(M) x 64(N).
// Each wave computes one 16x64 strip (4 accumulators) -> A fragment reused x4.
// The 64x32 f16 weight tile is staged in LDS (shared by all 8 waves),
// double-buffered via async global->LDS copies overlapping the WMMAs.
// K must be a multiple of 32; N a multiple of 64; M a multiple of 128.
#define GEMM_BN 64
#define GEMM_KB 32
#define LDS_PITCH 40   // 32 halfs + 8 pad (spreads banks for fragment reads)

// Load all 4 B fragments first, then fire 4 back-to-back WMMAs (single dscnt wait,
// clean XDL pipelining), A fragment reused across all 4.
__device__ __forceinline__ void mm_strip(const _Float16* __restrict__ bufp,
                                         int ln16, int half, v16h av,
                                         v8f& a0, v8f& a1, v8f& a2, v8f& a3) {
  const _Float16* bp0 = bufp + (ln16     ) * LDS_PITCH + half * 8;
  const _Float16* bp1 = bufp + (ln16 + 16) * LDS_PITCH + half * 8;
  const _Float16* bp2 = bufp + (ln16 + 32) * LDS_PITCH + half * 8;
  const _Float16* bp3 = bufp + (ln16 + 48) * LDS_PITCH + half * 8;
  v16h bv0 = cat8(*(const v8h*)bp0, *(const v8h*)(bp0 + 16));
  v16h bv1 = cat8(*(const v8h*)bp1, *(const v8h*)(bp1 + 16));
  v16h bv2 = cat8(*(const v8h*)bp2, *(const v8h*)(bp2 + 16));
  v16h bv3 = cat8(*(const v8h*)bp3, *(const v8h*)(bp3 + 16));
  a0 = __builtin_amdgcn_wmma_f32_16x16x32_f16(false, av, false, bv0, (short)0, a0, false, false);
  a1 = __builtin_amdgcn_wmma_f32_16x16x32_f16(false, av, false, bv1, (short)0, a1, false, false);
  a2 = __builtin_amdgcn_wmma_f32_16x16x32_f16(false, av, false, bv2, (short)0, a2, false, false);
  a3 = __builtin_amdgcn_wmma_f32_16x16x32_f16(false, av, false, bv3, (short)0, a3, false, false);
}

__global__ void k_gemm_wmma(const _Float16* __restrict__ A, const _Float16* __restrict__ W,
                            float* __restrict__ C, int M, int N, int K) {
  __shared__ _Float16 sb[2][GEMM_BN * LDS_PITCH];
  const int t    = threadIdx.x;
  const int wave = t >> 5;
  const int lane = t & 31;
  const int half = lane >> 4;
  const int ln16 = lane & 15;
  const int tileM = blockIdx.y * 8 + wave;
  const int nbase = blockIdx.x * GEMM_BN;

  // Staging role: thread t owns one 16B segment: row sn (0..63), k-segment sseg (0..3)
  const int sn = t >> 2, sseg = t & 3;
  const _Float16* gB = W + (size_t)(nbase + sn) * K + sseg * 8;
  _Float16* lB0 = &sb[0][sn * LDS_PITCH + sseg * 8];
  _Float16* lB1 = &sb[1][sn * LDS_PITCH + sseg * 8];

  const _Float16* pA = A + (size_t)(tileM * 16 + ln16) * K;
  const int aoff = half * 8;

  v8f acc0 = {}, acc1 = {}, acc2 = {}, acc3 = {};
  const int nIter = K / GEMM_KB;

  stage_b128(gB, lB0);  // prologue: buffer 0 <- k-slice 0

  // Steady state: next-slice async copy + current-slice WMMAs, unconditionally.
  int it = 0;
  for (; it < nIter - 1; ++it) {
    const int kb = it * GEMM_KB;
    stage_b128(gB + kb + GEMM_KB, (it & 1) ? lB0 : lB1);

    __builtin_prefetch(pA + kb + 2 * GEMM_KB, 0, 1);
    v8h alo = *(const v8h*)(pA + kb + aoff);
    v8h ahi = *(const v8h*)(pA + kb + 16 + aoff);
    v16h av = cat8(alo, ahi);

    wait_async_le1();        // current buffer's (older) copy has landed
    __syncthreads();         // ... for every wave in the block
    mm_strip(sb[it & 1], ln16, half, av, acc0, acc1, acc2, acc3);
    __syncthreads();         // everyone done reading before this buffer is reused
  }

  // Epilogue: last k-slice, nothing left in flight afterwards.
  {
    const int kb = it * GEMM_KB;
    v8h alo = *(const v8h*)(pA + kb + aoff);
    v8h ahi = *(const v8h*)(pA + kb + 16 + aoff);
    v16h av = cat8(alo, ahi);
    wait_async_le0();
    __syncthreads();
    mm_strip(sb[it & 1], ln16, half, av, acc0, acc1, acc2, acc3);
  }

  // C/D layout: lane n=lane&15, VGPR r -> row m = r + 8*half
  float* pC = C + (size_t)(tileM * 16) * N + nbase + ln16;
  #pragma unroll
  for (int r = 0; r < 8; ++r) {
    const size_t ro = (size_t)(r + 8 * half) * N;
    pC[ro +  0] = acc0[r];
    pC[ro + 16] = acc1[r];
    pC[ro + 32] = acc2[r];
    pC[ro + 48] = acc3[r];
  }
}

// -------- causal depthwise conv (DC=2) + silu, xz -> u --------
__global__ void k_conv_silu(const float* __restrict__ xz, const float* __restrict__ cw,
                            const float* __restrict__ cb, float* __restrict__ u) {
  const int o  = blockIdx.x * 256 + threadIdx.x;  // over B*L*DI
  const int di = o & (DI_ - 1);
  const int bl = o >> 8;                          // b*L + l
  const int l  = bl & (L_ - 1);
  const float cur  = xz[(size_t)bl * (2 * DI_) + di];
  const float prev = (l == 0) ? 0.f : xz[(size_t)(bl - 1) * (2 * DI_) + di];
  const float a = fmaf(cw[di * DC_ + 0], prev, fmaf(cw[di * DC_ + 1], cur, cb[di]));
  u[o] = siluf_(a);
}

// -------- x_proj (N=20) + delta = softplus(dt @ dt_w^T + dt_b), fused per (b,l) --------
__global__ void k_xproj_delta(const float* __restrict__ u, const float* __restrict__ xpw,
                              const float* __restrict__ dtw, const float* __restrict__ dtb,
                              float* __restrict__ xdbl, float* __restrict__ delta) {
  __shared__ float su[DI_];
  __shared__ float sx[DTR_ + 2 * DS_];
  const int bl = blockIdx.x;   // b*L + l
  const int t  = threadIdx.x;  // 0..255
  su[t] = u[(size_t)bl * DI_ + t];
  __syncthreads();
  if (t < DTR_ + 2 * DS_) {
    const float* wr = xpw + (size_t)t * DI_;
    float acc = 0.f;
    #pragma unroll 4
    for (int d = 0; d < DI_; ++d) acc = fmaf(su[d], wr[d], acc);
    sx[t] = acc;
    xdbl[(size_t)bl * 20 + t] = acc;
  }
  __syncthreads();
  const float* dwr = dtw + (size_t)t * DTR_;
  float acc = dtb[t];
  #pragma unroll
  for (int r = 0; r < DTR_; ++r) acc = fmaf(sx[r], dwr[r], acc);
  delta[(size_t)bl * DI_ + t] = (acc > 20.f) ? acc : log1pf(__expf(acc));  // softplus
}

// -------- selective scan (DS=2) + C-proj + D skip + silu(z) gate, writes f16 --------
__global__ void k_scan(const float* __restrict__ delta, const float* __restrict__ xdbl,
                       const float* __restrict__ u, const float* __restrict__ xz,
                       const float* __restrict__ A_log, const float* __restrict__ Dp,
                       _Float16* __restrict__ y16) {
  const int b  = blockIdx.x;
  const int di = threadIdx.x;
  const float A0 = -__expf(A_log[di * DS_ + 0]);
  const float A1 = -__expf(A_log[di * DS_ + 1]);
  const float dpar = Dp[di];
  float h0 = 0.f, h1 = 0.f;
  for (int l = 0; l < L_; ++l) {
    const size_t bl = (size_t)b * L_ + l;
    const float dt = delta[bl * DI_ + di];
    const float uu = u[bl * DI_ + di];
    const float z  = xz[bl * (2 * DI_) + DI_ + di];
    const float* xd = xdbl + bl * 20;
    const float B0 = xd[16], B1 = xd[17], C0 = xd[18], C1 = xd[19];
    h0 = fmaf(__expf(dt * A0), h0, dt * B0 * uu);
    h1 = fmaf(__expf(dt * A1), h1, dt * B1 * uu);
    const float y = fmaf(h0, C0, fmaf(h1, C1, uu * dpar));
    y16[bl * DI_ + di] = (_Float16)(y * siluf_(z));
  }
}

// -------- final fc (CH=32) + sigmoid; 8 rows per block, one wave per row --------
__global__ void k_fc_sigmoid(const float* __restrict__ x, const float* __restrict__ fw,
                             const float* __restrict__ fb, float* __restrict__ out) {
  __shared__ float sx[8 * D_];
  const int rowbase = blockIdx.x * 8;
  const int t = threadIdx.x;
  #pragma unroll
  for (int i = 0; i < 8; ++i)
    sx[i * 256 + t] = x[(size_t)rowbase * D_ + i * 256 + t];
  __syncthreads();
  const int wave = t >> 5, lane = t & 31;     // lane == output channel
  const float* xr = sx + wave * D_;
  const float* wr = fw + lane * D_;
  float acc = fb[lane];
  #pragma unroll 4
  for (int d = 0; d < D_; ++d) acc = fmaf(xr[d], wr[d], acc);
  out[(size_t)(rowbase + wave) * CH_ + lane] = sigmoidf_(acc);
}

extern "C" void kernel_launch(void* const* d_in, const int* in_sizes, int n_in,
                              void* d_out, int out_size, void* d_ws, size_t ws_size,
                              hipStream_t stream) {
  const float* noise = (const float*)d_in[0];
  const float* l1w   = (const float*)d_in[1];
  const float* l1b   = (const float*)d_in[2];
  const float* pos   = (const float*)d_in[3];
  const float* lnw   = (const float*)d_in[4];
  const float* lnb   = (const float*)d_in[5];
  const float* inpw  = (const float*)d_in[6];
  const float* convw = (const float*)d_in[7];
  const float* convb = (const float*)d_in[8];
  const float* xpw   = (const float*)d_in[9];
  const float* dtw   = (const float*)d_in[10];
  const float* dtb   = (const float*)d_in[11];
  const float* alog  = (const float*)d_in[12];
  const float* dpar  = (const float*)d_in[13];
  const float* outw  = (const float*)d_in[14];
  const float* fcw   = (const float*)d_in[15];
  const float* fcb   = (const float*)d_in[16];
  float* outp = (float*)d_out;

  // Workspace layout (~409 MB total)
  char* ws = (char*)d_ws;
  const size_t BLD = (size_t)B_ * L_ * D_;           // 16,777,216
  float*    x_buf = (float*)(ws);                                  // BLD f32
  _Float16* h16   = (_Float16*)(ws + BLD * 4);                     // BLD f16
  float*    xz    = (float*)(ws + BLD * 4 + BLD * 2);              // 2*BLD f32
  float*    u_buf = (float*)((char*)xz    + BLD * 8);              // BLD f32
  float*    xdbl  = (float*)((char*)u_buf + BLD * 4);              // B*L*20 f32
  float*    delta = (float*)((char*)xdbl  + (size_t)B_ * L_ * 20 * 4); // BLD f32
  _Float16* y16   = (_Float16*)((char*)delta + BLD * 4);           // BLD f16
  _Float16* w16   = (_Float16*)((char*)y16   + BLD * 2);           // 512*256 f16 (reused)

  // x = noise @ l1_w^T + l1_b + pos
  k_init<<<dim3(L_ * D_ / 256, B_), 256, 0, stream>>>(noise, l1w, l1b, pos, x_buf);

  for (int li = 0; li < NL_; ++li) {
    k_layernorm<<<(B_ * L_) / 8, 256, 0, stream>>>(x_buf, lnw + li * D_, lnb + li * D_, h16);

    const int nin = 2 * DI_ * D_;
    k_cvt_f16<<<(nin + 255) / 256, 256, 0, stream>>>(inpw + (size_t)li * nin, w16, nin);
    // xz = h @ in_proj_w^T   (M=65536, N=512, K=256)
    k_gemm_wmma<<<dim3((2 * DI_) / GEMM_BN, (B_ * L_) / 128), 256, 0, stream>>>(
        h16, w16, xz, B_ * L_, 2 * DI_, D_);

    k_conv_silu<<<(B_ * L_ * DI_) / 256, 256, 0, stream>>>(
        xz, convw + li * DI_ * DC_, convb + li * DI_, u_buf);

    k_xproj_delta<<<B_ * L_, 256, 0, stream>>>(
        u_buf, xpw + (size_t)li * (DTR_ + 2 * DS_) * DI_,
        dtw + (size_t)li * DI_ * DTR_, dtb + li * DI_, xdbl, delta);

    k_scan<<<B_, DI_, 0, stream>>>(delta, xdbl, u_buf, xz,
                                   alog + li * DI_ * DS_, dpar + li * DI_, y16);

    const int nout = D_ * DI_;
    k_cvt_f16<<<(nout + 255) / 256, 256, 0, stream>>>(outw + (size_t)li * nout, w16, nout);
    // x = y @ out_proj_w^T   (M=65536, N=256, K=256)
    k_gemm_wmma<<<dim3(D_ / GEMM_BN, (B_ * L_) / 128), 256, 0, stream>>>(
        y16, w16, x_buf, B_ * L_, D_, DI_);
  }

  k_fc_sigmoid<<<(B_ * L_) / 8, 256, 0, stream>>>(x_buf, fcw, fcb, outp);

  (void)in_sizes; (void)n_in; (void)out_size; (void)ws_size;
}
